// Encoder_45827301048610
// MI455X (gfx1250) — compile-verified, matrix operand order checked by reading, not run on hardware
//
#include <hip/hip_runtime.h>
#include <hip/hip_bf16.h>

// ---------------------------------------------------------------------------
// Problem constants (from the reference): D=4 layers, L=16 frames,
// C=64 channels, H=W=128.
// ---------------------------------------------------------------------------
#define Dd 4
#define Ld 16
#define Cd 64
#define Hd 128
#define Wd 128
#define HW (Hd * Wd)        /* 16384  */
#define CHW (Cd * HW)       /* 1048576 */
#define LCHW (Ld * CHW)     /* 16777216 */

typedef __attribute__((ext_vector_type(16))) __bf16 v16bf;
typedef __attribute__((ext_vector_type(8)))  float  v8f;

// D = A(16x32 bf16) * B(32x16 bf16) + C(16x16 f32), wave32.
__device__ __forceinline__ v8f wmma_bf16(v16bf a, v16bf b, v8f c) {
  return __builtin_amdgcn_wmma_f32_16x16x32_bf16(
      /*neg_a=*/false, a, /*neg_b=*/false, b,
      /*c_mod=*/(short)0, c, /*reuse_a=*/false, /*reuse_b=*/false);
}

// 16-bit A/B operand striping (CDNA5 ISA 7.12.2): element e of the v16bf
// held by a lane maps to K index:
//   lane 0-15 : e<8 -> K=e       ; e>=8 -> K=e+8   (K 0-7, 16-23)
//   lane16-31 : e<8 -> K=e+8     ; e>=8 -> K=e+16  (K 8-15, 24-31)
__device__ __forceinline__ int kk16(int e, int hi) {
  return ((e < 8) ? e : e + 8) + (hi ? 8 : 0);
}

// --- CDNA5 async global->LDS copy (ASYNCcnt path, ISA ch.10 / 15.18) -------
// 16B per lane: LDS[lds_off] = MEM[gaddr]. Tracked by ASYNCcnt.
__device__ __forceinline__ void async_g2lds_b128(unsigned lds_off,
                                                 const void* gaddr) {
  unsigned long long ga = (unsigned long long)(uintptr_t)gaddr;
  asm volatile("global_load_async_to_lds_b128 %0, %1, off"
               :: "v"(lds_off), "v"(ga)
               : "memory");
}
__device__ __forceinline__ void wait_async0() {
  asm volatile("s_wait_asynccnt 0" ::: "memory");
}
// Generic shared-pointer low 32 bits == LDS byte offset (ISA 10.2 aperture).
__device__ __forceinline__ unsigned lds_off32(const void* p) {
  return (unsigned)(uintptr_t)p;
}

// ---------------------------------------------------------------------------
// LayerNorm statistics, direct form (used once, for layer-0 ln1):
// one block per frame, reduce mean / rsqrt(var+eps) over C*H*W elements.
// ---------------------------------------------------------------------------
__global__ void __launch_bounds__(256)
ln_stats_kernel(const float* __restrict__ x, float* __restrict__ stats) {
  __shared__ float s_sum[256];
  __shared__ float s_sq[256];
  const int l = blockIdx.x;
  const float* xf = x + (size_t)l * CHW;
  float s = 0.f, q = 0.f;
  for (int i = threadIdx.x; i < CHW; i += 256) {
    float v = xf[i];
    s += v;
    q = fmaf(v, v, q);
  }
  s_sum[threadIdx.x] = s;
  s_sq[threadIdx.x] = q;
  __syncthreads();
  for (int ofs = 128; ofs > 0; ofs >>= 1) {
    if (threadIdx.x < (unsigned)ofs) {
      s_sum[threadIdx.x] += s_sum[threadIdx.x + ofs];
      s_sq[threadIdx.x]  += s_sq[threadIdx.x + ofs];
    }
    __syncthreads();
  }
  if (threadIdx.x == 0) {
    float mu  = s_sum[0] * (1.0f / CHW);
    float var = s_sq[0] * (1.0f / CHW) - mu * mu;
    stats[2 * l]     = mu;
    stats[2 * l + 1] = rsqrtf(var + 1e-5f);
  }
}

// Finalize fused LN stats: partials[b] = (sum, sumsq) per producing block,
// bpf blocks per frame, deterministic tree reduction (no float atomics).
__global__ void __launch_bounds__(256)
finalize_stats_kernel(const float* __restrict__ part, int bpf,
                      float* __restrict__ stats) {
  __shared__ float s_sum[256];
  __shared__ float s_sq[256];
  const int l = blockIdx.x;
  float s = 0.f, q = 0.f;
  for (int b = threadIdx.x; b < bpf; b += 256) {
    s += part[(l * bpf + b) * 2 + 0];
    q += part[(l * bpf + b) * 2 + 1];
  }
  s_sum[threadIdx.x] = s;
  s_sq[threadIdx.x] = q;
  __syncthreads();
  for (int ofs = 128; ofs > 0; ofs >>= 1) {
    if (threadIdx.x < (unsigned)ofs) {
      s_sum[threadIdx.x] += s_sum[threadIdx.x + ofs];
      s_sq[threadIdx.x]  += s_sq[threadIdx.x + ofs];
    }
    __syncthreads();
  }
  if (threadIdx.x == 0) {
    float mu  = s_sum[0] * (1.0f / CHW);
    float var = s_sq[0] * (1.0f / CHW) - mu * mu;
    stats[2 * l]     = mu;
    stats[2 * l + 1] = rsqrtf(var + 1e-5f);
  }
}

// ---------------------------------------------------------------------------
// Fused LN1-apply + Q/K/V 1x1 convs (per-pixel GEMM over channels) via WMMA.
// Weights are async-copied raw (f32) into LDS so the copy overlaps the
// activation (B-tile) global loads; bf16 conversion happens at A-tile build.
// Each wave owns one 16-pixel tile. grid = (L*HW/16)/8 blocks x 256 threads.
// ---------------------------------------------------------------------------
__global__ void __launch_bounds__(256)
qkv_kernel(const float* __restrict__ x, const float* __restrict__ g,
           const float* __restrict__ bln, const float* __restrict__ stats,
           const float* __restrict__ Wq, const float* __restrict__ bq,
           const float* __restrict__ Wk, const float* __restrict__ bk,
           const float* __restrict__ Wv, const float* __restrict__ bv,
           float* __restrict__ Q, float* __restrict__ Kp,
           float* __restrict__ Vp) {
  __shared__ __align__(16) float sW32[3 * 64 * 64];   // 48 KB, f32 weights
  __shared__ float sB[3][64];

  // Issue async weight copies first (ASYNCcnt), consume much later.
  {
    const unsigned lbase = lds_off32(&sW32[0]);
    const float* srcs[3] = {Wq, Wk, Wv};
#pragma unroll
    for (int m = 0; m < 3; ++m) {
      for (int i = threadIdx.x; i < 1024; i += 256)   // vec4 units
        async_g2lds_b128(lbase + (unsigned)(m * 4096 + i * 4) * 4u,
                         srcs[m] + i * 4);
    }
    if (threadIdx.x < 64) {
      sB[0][threadIdx.x] = bq[threadIdx.x];
      sB[1][threadIdx.x] = bk[threadIdx.x];
      sB[2][threadIdx.x] = bv[threadIdx.x];
    }
  }

  const int wave = threadIdx.x >> 5;
  const int lane = threadIdx.x & 31;
  const int hi = lane >> 4;       // lane half
  const int n  = lane & 15;       // B/D column (pixel within tile), A row
  const int tile = blockIdx.x * 8 + wave;       // 16 consecutive pixels
  const int l  = tile >> 10;                    // HW/16 = 1024 tiles/frame
  const int q0 = (tile & 1023) << 4;
  const float mu = stats[2 * l];
  const float rs = stats[2 * l + 1];
  const float* xl = x + (size_t)l * CHW;

  // LN-applied activation tile in B layout, two K=32 chunks (overlaps the
  // in-flight async weight copies).
  v16bf bt[2];
#pragma unroll
  for (int kc = 0; kc < 2; ++kc) {
#pragma unroll
    for (int e = 0; e < 16; ++e) {
      int ci  = 32 * kc + kk16(e, hi);
      int idx = ci * HW + q0 + n;
      float v = (xl[idx] - mu) * rs;
      v = fmaf(v, g[idx], bln[idx]);
      bt[kc][e] = (__bf16)v;
    }
  }

  wait_async0();       // my async copies landed in LDS
  __syncthreads();     // everyone's copies landed

  float* outs[3] = {Q, Kp, Vp};
#pragma unroll
  for (int m = 0; m < 3; ++m) {
#pragma unroll
    for (int t = 0; t < 4; ++t) {      // 4 output-channel tiles of 16
      v8f acc = {};
#pragma unroll
      for (int kc = 0; kc < 2; ++kc) {
        const float* wb = sW32 + m * 4096 + (16 * t + n) * 64 + 32 * kc;
        v16bf at;
#pragma unroll
        for (int e = 0; e < 16; e += 2) {   // K pairs are contiguous
          int k0 = kk16(e, hi);
          at[e]     = (__bf16)wb[k0];
          at[e + 1] = (__bf16)wb[k0 + 1];
        }
        acc = wmma_bf16(at, bt[kc], acc);
      }
      float* op = outs[m] + (size_t)l * CHW + q0 + n;
#pragma unroll
      for (int r = 0; r < 8; ++r) {    // D: lane half selects M rows r / r+8
        int co = 16 * t + r + 8 * hi;
        op[(size_t)co * HW] = acc[r] + sB[m][co];
      }
    }
  }
}

// ---------------------------------------------------------------------------
// Split 5x5 attention conv: A = conv5(Q, Wa[:, :C]), B = conv5(K, Wa[:, C:]).
// ba[d] is a uniform additive constant before softmax => softmax-invariant
// => dropped. One thread per output pixel; Wa async-copied into LDS.
// ---------------------------------------------------------------------------
__global__ void __launch_bounds__(256)
conv5_kernel(const float* __restrict__ Q, const float* __restrict__ Kt,
             const float* __restrict__ Wa, float* __restrict__ Aat,
             float* __restrict__ Bat) {
  __shared__ __align__(16) float sWa[2 * 64 * 25];   // 12.8 KB
  {
    const unsigned lbase = lds_off32(&sWa[0]);
    for (int i = threadIdx.x; i < (2 * 64 * 25) / 4; i += 256)  // 800 vec4s
      async_g2lds_b128(lbase + (unsigned)i * 16u, Wa + i * 4);
  }

  const int idx = blockIdx.x * 256 + threadIdx.x;   // over L*HW
  const int l = idx >> 14;
  const int q = idx & (HW - 1);
  const int y = q >> 7;
  const int xx = q & (Wd - 1);
  const float* Ql = Q + (size_t)l * CHW;
  const float* Kl = Kt + (size_t)l * CHW;

  wait_async0();
  __syncthreads();

  float accA = 0.f, accB = 0.f;
  for (int ci = 0; ci < 64; ++ci) {
    const float* qc = Ql + (size_t)ci * HW;
    const float* kc = Kl + (size_t)ci * HW;
    __builtin_prefetch(qc + HW + q, 0, 0);   // next channel plane
    __builtin_prefetch(kc + HW + q, 0, 0);
    const float* wq = sWa + ci * 25;
    const float* wk = sWa + (64 + ci) * 25;
#pragma unroll
    for (int dy = 0; dy < 5; ++dy) {
      int yy = y + dy - 2;
      if (yy < 0 || yy >= Hd) continue;
#pragma unroll
      for (int dx = 0; dx < 5; ++dx) {
        int x2 = xx + dx - 2;
        if (x2 < 0 || x2 >= Wd) continue;
        accA = fmaf(wq[dy * 5 + dx], qc[yy * Wd + x2], accA);
        accB = fmaf(wk[dy * 5 + dx], kc[yy * Wd + x2], accB);
      }
    }
  }
  Aat[idx] = accA;
  Bat[idx] = accB;
}

// ---------------------------------------------------------------------------
// Per-pixel frame softmax + attn = P.V + residual (in place on x), fused
// with ln2-stat partial reduction of the freshly written x (saves a full
// 67MB re-read of x at 23.3 TB/s). One thread per (target frame i, pixel q).
// ---------------------------------------------------------------------------
__global__ void __launch_bounds__(256)
attn_kernel(const float* __restrict__ Aat, const float* __restrict__ Bat,
            const float* __restrict__ V, float* __restrict__ x,
            float* __restrict__ part) {
  __shared__ float s_sum[256];
  __shared__ float s_sq[256];
  const int idx = blockIdx.x * 256 + threadIdx.x;   // over L*HW
  const int i = idx >> 14;
  const int q = idx & (HW - 1);
  const float a = Aat[idx];
  float p[Ld];
  float mx = -3.0e38f;
#pragma unroll
  for (int j = 0; j < Ld; ++j) {
    p[j] = a + Bat[j * HW + q];
    mx = fmaxf(mx, p[j]);
  }
  float s = 0.f;
#pragma unroll
  for (int j = 0; j < Ld; ++j) {
    p[j] = __expf(p[j] - mx);
    s += p[j];
  }
  const float inv = 1.0f / s;
#pragma unroll
  for (int j = 0; j < Ld; ++j) p[j] *= inv;

  float* xo = x + (size_t)i * CHW + q;
  float ts = 0.f, tq = 0.f;
#pragma unroll 4
  for (int c = 0; c < Cd; ++c) {
    const float* vp = V + (size_t)c * HW + q;
    float acc = 0.f;
#pragma unroll
    for (int j = 0; j < Ld; ++j) acc = fmaf(p[j], vp[(size_t)j * CHW], acc);
    float nv = xo[(size_t)c * HW] + acc;
    xo[(size_t)c * HW] = nv;
    ts += nv;
    tq = fmaf(nv, nv, tq);
  }

  // Block-level (sum, sumsq) partial for ln2 (block spans one frame).
  s_sum[threadIdx.x] = ts;
  s_sq[threadIdx.x] = tq;
  __syncthreads();
  for (int ofs = 128; ofs > 0; ofs >>= 1) {
    if (threadIdx.x < (unsigned)ofs) {
      s_sum[threadIdx.x] += s_sum[threadIdx.x + ofs];
      s_sq[threadIdx.x]  += s_sq[threadIdx.x + ofs];
    }
    __syncthreads();
  }
  if (threadIdx.x == 0) {
    part[blockIdx.x * 2 + 0] = s_sum[0];
    part[blockIdx.x * 2 + 1] = s_sq[0];
  }
}

// ---------------------------------------------------------------------------
// Fused FFN: LN2-apply -> GEMM1(128x64) -> bias+LeakyReLU -> lane-local
// repack of f32 D-tiles into bf16 B-layout chunks -> GEMM2(64x128) ->
// bias + residual in place, fused with next-layer ln1 stat partials.
// ---------------------------------------------------------------------------
__global__ void __launch_bounds__(256)
ffn_kernel(const float* __restrict__ g, const float* __restrict__ bln,
           const float* __restrict__ stats,
           const float* __restrict__ W1, const float* __restrict__ b1,
           const float* __restrict__ W2, const float* __restrict__ b2,
           float* __restrict__ x, float* __restrict__ part) {
  __shared__ __bf16 sW1[128 * 64];   // 16 KB
  __shared__ __bf16 sW2[64 * 128];   // 16 KB
  __shared__ float  sb1[128];
  __shared__ float  sb2[64];
  __shared__ float  s_sum[256];
  __shared__ float  s_sq[256];
  for (int i = threadIdx.x; i < 128 * 64; i += 256) {
    sW1[i] = (__bf16)W1[i];
    sW2[i] = (__bf16)W2[i];
  }
  if (threadIdx.x < 128) sb1[threadIdx.x] = b1[threadIdx.x];
  if (threadIdx.x < 64)  sb2[threadIdx.x] = b2[threadIdx.x];
  __syncthreads();

  const int wave = threadIdx.x >> 5;
  const int lane = threadIdx.x & 31;
  const int hi = lane >> 4;
  const int n  = lane & 15;
  const int tile = blockIdx.x * 8 + wave;
  const int l  = tile >> 10;
  const int q0 = (tile & 1023) << 4;
  const float mu = stats[2 * l];
  const float rs = stats[2 * l + 1];
  float* xl = x + (size_t)l * CHW;

  // LN2-applied activation tile, B layout, two K=32 chunks.
  v16bf bt[2];
#pragma unroll
  for (int kc = 0; kc < 2; ++kc) {
#pragma unroll
    for (int e = 0; e < 16; ++e) {
      int ci  = 32 * kc + kk16(e, hi);
      int idx = ci * HW + q0 + n;
      float v = (xl[idx] - mu) * rs;
      v = fmaf(v, g[idx], bln[idx]);
      bt[kc][e] = (__bf16)v;
    }
  }

  // GEMM1: f1[m] for m in 0..127 -> 8 D-tiles, + bias + leaky_relu(0.01).
  float f1v[8][8];
#pragma unroll
  for (int t = 0; t < 8; ++t) {
    v8f acc = {};
#pragma unroll
    for (int kc = 0; kc < 2; ++kc) {
      v16bf at;
#pragma unroll
      for (int e = 0; e < 16; ++e)
        at[e] = sW1[(16 * t + n) * 64 + 32 * kc + kk16(e, hi)];
      acc = wmma_bf16(at, bt[kc], acc);
    }
#pragma unroll
    for (int r = 0; r < 8; ++r) {
      int co = 16 * t + r + 8 * hi;
      float v = acc[r] + sb1[co];
      f1v[t][r] = (v > 0.f) ? v : 0.01f * v;
    }
  }

  // Repack D-layout f1 tiles into 4 bf16 B-layout K=32 chunks (lane-local:
  // D (lane=N, vgpr=M-row) lines up with B chunk-by-chunk, no shuffles).
  v16bf fb[4];
#pragma unroll
  for (int kc = 0; kc < 4; ++kc) {
#pragma unroll
    for (int e = 0; e < 16; ++e)
      fb[kc][e] = (__bf16)f1v[2 * kc + (e >> 3)][e & 7];
  }

  // GEMM2: out[co] = W2[co, :128] . f1 + b2 + residual; track (sum, sumsq).
  float ts = 0.f, tq = 0.f;
#pragma unroll
  for (int t = 0; t < 4; ++t) {
    v8f acc = {};
#pragma unroll
    for (int kc = 0; kc < 4; ++kc) {
      v16bf at;
#pragma unroll
      for (int e = 0; e < 16; ++e)
        at[e] = sW2[(16 * t + n) * 128 + 32 * kc + kk16(e, hi)];
      acc = wmma_bf16(at, fb[kc], acc);
    }
    float* op = xl + q0 + n;
#pragma unroll
    for (int r = 0; r < 8; ++r) {
      int co = 16 * t + r + 8 * hi;
      float nv = op[(size_t)co * HW] + acc[r] + sb2[co];
      op[(size_t)co * HW] = nv;
      ts += nv;
      tq = fmaf(nv, nv, tq);
    }
  }

  // Block-level (sum, sumsq) partial for next layer's ln1 (block = 1 frame).
  s_sum[threadIdx.x] = ts;
  s_sq[threadIdx.x] = tq;
  __syncthreads();
  for (int ofs = 128; ofs > 0; ofs >>= 1) {
    if (threadIdx.x < (unsigned)ofs) {
      s_sum[threadIdx.x] += s_sum[threadIdx.x + ofs];
      s_sq[threadIdx.x]  += s_sq[threadIdx.x + ofs];
    }
    __syncthreads();
  }
  if (threadIdx.x == 0) {
    part[blockIdx.x * 2 + 0] = s_sum[0];
    part[blockIdx.x * 2 + 1] = s_sq[0];
  }
}

// ---------------------------------------------------------------------------
// Host-side launcher. Scratch carved out of d_ws:
//   xbuf | Q | K | V : LCHW floats,  Aat/Bat : L*HW,  stats : 2*L,
//   partials : 2048*2.  Input x is never mutated; all ops on `stream`.
// ---------------------------------------------------------------------------
extern "C" void kernel_launch(void* const* d_in, const int* in_sizes, int n_in,
                              void* d_out, int out_size, void* d_ws,
                              size_t ws_size, hipStream_t stream) {
  const float* x_in  = (const float*)d_in[0];
  const float* ln1_g = (const float*)d_in[1];
  const float* ln1_b = (const float*)d_in[2];
  const float* Wq = (const float*)d_in[3];
  const float* bq = (const float*)d_in[4];
  const float* Wk = (const float*)d_in[5];
  const float* bk = (const float*)d_in[6];
  const float* Wv = (const float*)d_in[7];
  const float* bv = (const float*)d_in[8];
  const float* Wa = (const float*)d_in[9];
  // d_in[10] = ba: uniform pre-softmax shift, softmax-invariant -> unused.
  const float* ln2_g = (const float*)d_in[11];
  const float* ln2_b = (const float*)d_in[12];
  const float* W1 = (const float*)d_in[13];
  const float* b1 = (const float*)d_in[14];
  const float* W2 = (const float*)d_in[15];
  const float* b2 = (const float*)d_in[16];

  float* xbuf  = (float*)d_ws;
  float* Qb    = xbuf + (size_t)LCHW;
  float* Kb    = Qb + (size_t)LCHW;
  float* Vb    = Kb + (size_t)LCHW;
  float* Aat   = Vb + (size_t)LCHW;
  float* Bat   = Aat + (size_t)Ld * HW;
  float* stats = Bat + (size_t)Ld * HW;
  float* parts = stats + 2 * Ld;

  hipMemcpyAsync(xbuf, x_in, sizeof(float) * (size_t)LCHW,
                 hipMemcpyDeviceToDevice, stream);

  const int gemm_blocks = (Ld * HW / 16) / 8;   // 2048 (8 waves/block)
  const int pix_blocks  = (Ld * HW) / 256;      // 1024

  for (int d = 0; d < Dd; ++d) {
    if (d == 0)   // later layers get ln1 stats fused into the previous FFN
      ln_stats_kernel<<<Ld, 256, 0, stream>>>(xbuf, stats);
    qkv_kernel<<<gemm_blocks, 256, 0, stream>>>(
        xbuf, ln1_g + (size_t)d * CHW, ln1_b + (size_t)d * CHW, stats,
        Wq + (size_t)d * Cd * Cd, bq + (size_t)d * Cd,
        Wk + (size_t)d * Cd * Cd, bk + (size_t)d * Cd,
        Wv + (size_t)d * Cd * Cd, bv + (size_t)d * Cd, Qb, Kb, Vb);
    conv5_kernel<<<pix_blocks, 256, 0, stream>>>(
        Qb, Kb, Wa + (size_t)d * 2 * Cd * 25, Aat, Bat);
    attn_kernel<<<pix_blocks, 256, 0, stream>>>(Aat, Bat, Vb, xbuf, parts);
    finalize_stats_kernel<<<Ld, 256, 0, stream>>>(
        parts, pix_blocks / Ld, stats);                  // ln2 stats
    ffn_kernel<<<gemm_blocks, 256, 0, stream>>>(
        ln2_g + (size_t)d * CHW, ln2_b + (size_t)d * CHW, stats,
        W1 + (size_t)d * 2 * Cd * Cd, b1 + (size_t)d * 2 * Cd,
        W2 + (size_t)d * Cd * 2 * Cd, b2 + (size_t)d * Cd, xbuf, parts);
    finalize_stats_kernel<<<Ld, 256, 0, stream>>>(
        parts, gemm_blocks / Ld, stats);                 // next ln1 stats
  }

  hipMemcpyAsync(d_out, xbuf, sizeof(float) * (size_t)LCHW,
                 hipMemcpyDeviceToDevice, stream);
}
